// Attention_2259152797919
// MI455X (gfx1250) — matrix operand order repaired and checked by reading, not compiled, not run on hardware
//
#include <hip/hip_runtime.h>
#include <hip/hip_bf16.h>

// ---------------- CDNA5 WMMA / TDM types ----------------
typedef __attribute__((ext_vector_type(16))) __bf16 v16bf;
typedef __attribute__((ext_vector_type(8)))  float  v8f;
typedef __attribute__((ext_vector_type(4)))  unsigned int v4u;
typedef __attribute__((ext_vector_type(8)))  int v8i;
typedef __attribute__((ext_vector_type(4)))  int v4i;

union Frag16 { v16bf v; unsigned int u[8]; };

// float -> bf16 (round to nearest even), packed pair helpers
__device__ __forceinline__ unsigned short f2bf(float f) {
  unsigned int u = __builtin_bit_cast(unsigned int, f);
  u += 0x7FFFu + ((u >> 16) & 1u);
  return (unsigned short)(u >> 16);
}
__device__ __forceinline__ unsigned int pack_bf2(float lo, float hi) {
  return (unsigned int)f2bf(lo) | ((unsigned int)f2bf(hi) << 16);
}

// width-16 reductions (wave32; xor masks <16 stay within each 16-lane half)
__device__ __forceinline__ float hmax16(float x) {
  x = fmaxf(x, __shfl_xor(x, 1, 32));
  x = fmaxf(x, __shfl_xor(x, 2, 32));
  x = fmaxf(x, __shfl_xor(x, 4, 32));
  x = fmaxf(x, __shfl_xor(x, 8, 32));
  return x;
}
__device__ __forceinline__ float hsum16(float x) {
  x += __shfl_xor(x, 1, 32);
  x += __shfl_xor(x, 2, 32);
  x += __shfl_xor(x, 4, 32);
  x += __shfl_xor(x, 8, 32);
  return x;
}

#define NWIN   625   // 25*25
#define NPAD   640   // padded to 40 tiles of 16
#define DMODEL 256
#define NHEADS 8
#define DHEAD  32

// ---------------------------------------------------------------------
// Tensor Data Mover: 2D f32 tile load  global -> LDS
//   D# group0: count=1 | lds_addr | global_addr(57b) | type=2
//   D# group1: data_size=4B, tensor_dim0/1, tile_dim0/1, dim0_stride
// (5-arg builtin on ROCm7.2/clang-22, 6-arg on clang>=23)
// ---------------------------------------------------------------------
__device__ __forceinline__ void tdm_load_2d_f32(unsigned lds_byte_off,
                                                const float* gptr,
                                                unsigned tile_x, unsigned tile_y,
                                                unsigned tensor_x, unsigned tensor_y,
                                                unsigned stride_elems) {
  unsigned long long ga = (unsigned long long)(const void*)gptr;
  v4u g0;
  g0.x = 0x1u;                                   // count=1, user descriptor
  g0.y = lds_byte_off;                           // LDS byte address
  g0.z = (unsigned)ga;                           // global_addr[31:0]
  g0.w = (unsigned)(ga >> 32) | (2u << 30);      // global_addr[56:32] | type=2
  v8i g1;
  g1[0] = (int)(2u << 16);                       // data_size = 4 bytes
  g1[1] = (int)((tensor_x & 0xFFFFu) << 16);     // tensor_dim0[15:0]
  g1[2] = (int)((tensor_x >> 16) | ((tensor_y & 0xFFFFu) << 16)); // dim0 hi | dim1 lo
  g1[3] = (int)((tensor_y >> 16) | (tile_x << 16));               // dim1 hi | tile_dim0
  g1[4] = (int)(tile_y & 0xFFFFu);               // tile_dim1 (tile_dim2 = 0)
  g1[5] = (int)stride_elems;                     // tensor_dim0_stride[31:0]
  g1[6] = 0;
  g1[7] = 0;
  v4i z4 = {0, 0, 0, 0};
#if __clang_major__ >= 23
  v8i z8 = {0, 0, 0, 0, 0, 0, 0, 0};
  __builtin_amdgcn_tensor_load_to_lds(g0, g1, z4, z4, z8, 0);
#else
  __builtin_amdgcn_tensor_load_to_lds(g0, g1, z4, z4, 0);
#endif
}

// =====================================================================
// Kernel 1: fused transpose + QKV projection (bf16 WMMA, f32 accum)
// =====================================================================
__global__ __launch_bounds__(32)
void qkv_kernel(const float* __restrict__ x, const float* __restrict__ w,
                unsigned short* __restrict__ q, unsigned short* __restrict__ k,
                unsigned short* __restrict__ vT) {
  const int lane = threadIdx.x;
  const int ct = blockIdx.x;            // 0..47  (col tile of 768)
  const int nt = blockIdx.y;            // 0..39  (row tile of 640)
  const int b  = blockIdx.z;            // 0..63
  const int n0 = nt * 16, c0 = ct * 16;
  const int m    = lane & 15;
  const int half = lane >> 4;
  const int nn   = lane & 15;

  const int  nrow  = n0 + m;
  const bool rowOK = (nrow < NWIN);
  const int  nrs   = rowOK ? nrow : 0;           // clamped (always-safe) address
  const float wm   = rowOK ? 1.0f : 0.0f;        // branch-free zero-pad mask
  const float colscale = (c0 < DMODEL) ? 0.17677669529663687f : 1.0f; // 32^-0.5 for Q

  v8f acc = {};
#pragma unroll 2
  for (int k0 = 0; k0 < DMODEL; k0 += 32) {
    Frag16 a, bm;
#pragma unroll
    for (int vi = 0; vi < 8; ++vi) {
      // A fragment: xf[n][dd] = x[b][dd][n]   (K = dd, strided in memory)
      const int kbA = ((vi & 4) ? 16 : 0) + (half ? 8 : 0) + (vi & 3) * 2;
      const float* px = x + ((size_t)b * DMODEL + (k0 + kbA)) * NWIN + nrs;
      a.u[vi] = pack_bf2(px[0] * wm, px[NWIN] * wm);
      // B fragment: w_qkv[k][c]  (K rows stride 768)
      const int kbB = ((vi & 4) ? 16 : 0) + (half ? 8 : 0) + (vi & 3) * 2;
      const float* pw = w + (size_t)(k0 + kbB) * 768 + (c0 + nn);
      bm.u[vi] = pack_bf2(pw[0] * colscale, pw[768] * colscale);
    }
    acc = __builtin_amdgcn_wmma_f32_16x16x32_bf16(false, a.v, false, bm.v,
                                                  (short)0, acc, false, false);
  }

  // C layout: VGPR r -> row = r + 8*half, col = lane%16.
  // Region routing is WAVE-UNIFORM: a 16-col tile never straddles the
  // q/k/v boundaries (multiples of 256) -> branch on scalar c0, not lane c.
  const int c   = c0 + nn;
  const int hh  = (c & 255) >> 5;                // uniform across tile
  const int dhc = c & 31;
  const size_t bh = (size_t)b * NHEADS + hh;
  if (c0 < DMODEL) {
#pragma unroll
    for (int r = 0; r < 8; ++r) {
      const int rowg = n0 + r + 8 * half;
      q[(bh * NPAD + rowg) * DHEAD + dhc] = f2bf(acc[r]);
    }
  } else if (c0 < 2 * DMODEL) {
#pragma unroll
    for (int r = 0; r < 8; ++r) {
      const int rowg = n0 + r + 8 * half;
      k[(bh * NPAD + rowg) * DHEAD + dhc] = f2bf(acc[r]);
    }
  } else {
#pragma unroll
    for (int r = 0; r < 8; ++r) {
      const int rowg = n0 + r + 8 * half;
      vT[(bh * DHEAD + dhc) * NPAD + rowg] = f2bf(acc[r]);
    }
  }
}

// =====================================================================
// Kernel 2: materialize padded rel-pos bias (8, 640, 640) f32
// =====================================================================
__global__ __launch_bounds__(256)
void bias_kernel(const float* __restrict__ bt, const int* __restrict__ rpi,
                 float* __restrict__ biasPad) {
  const int idx = blockIdx.x * 256 + threadIdx.x;
  const int total = NHEADS * NPAD * NPAD;
  if (idx >= total) return;
  const int j = idx % NPAD;
  const int t = idx / NPAD;
  const int i = t % NPAD;
  const int h = t / NPAD;
  float v;
  if (j >= NWIN)      v = -1e30f;
  else if (i >= NWIN) v = 0.0f;
  else                v = bt[rpi[i * NWIN + j] * NHEADS + h];
  biasPad[idx] = v;
}

// =====================================================================
// Kernel 3: fused flash-attention per (b, h, 16-row tile)
//   bias tiles streamed by the Tensor Data Mover (double-buffered LDS)
// =====================================================================
__global__ __launch_bounds__(32)
void attn_kernel(const unsigned short* __restrict__ q,
                 const unsigned short* __restrict__ k,
                 const unsigned short* __restrict__ vT,
                 const float* __restrict__ biasPad,
                 unsigned short* __restrict__ attnout) {
  // single LDS object => known byte offsets for the TDM descriptor:
  //   [0,2048)    bias tile buffer 0 (16x32 f32)
  //   [2048,4096) bias tile buffer 1
  //   [4096,5120) P repack tile (16x32 bf16)
  __shared__ __align__(16) unsigned int ldsAll[1280];

  const int lane = threadIdx.x;
  const int it = blockIdx.x;            // 0..39
  const int h  = blockIdx.y;            // 0..7
  const int b  = blockIdx.z;            // 0..63
  const int i0 = it * 16;
  const int m    = lane & 15;
  const int half = lane >> 4;
  const int nn   = lane & 15;
  const size_t bh = (size_t)b * NHEADS + h;
  const float* biasBase = biasPad + ((size_t)h * NPAD + i0) * NPAD;  // 16x640 slab

  // A fragment of Q rows i0..i0+15 (loaded once; contiguous bf16 pairs)
  Frag16 aq;
#pragma unroll
  for (int vi = 0; vi < 8; ++vi) {
    const int kb = ((vi & 4) ? 16 : 0) + (half ? 8 : 0) + (vi & 3) * 2;
    aq.u[vi] = *(const unsigned int*)(q + (bh * NPAD + i0 + m) * DHEAD + kb);
  }

  float mrow[8], lrow[8];
  v8f acc0 = {}, acc1 = {};
#pragma unroll
  for (int r = 0; r < 8; ++r) { mrow[r] = -3e38f; lrow[r] = 0.0f; }

  // prologue: TDM the first 16x32 bias tile into buffer 0
  tdm_load_2d_f32(0u, biasBase, 32, 16, NPAD, NPAD, NPAD);

  for (int j0 = 0; j0 < NPAD; j0 += 32) {
    const int cur = (j0 >> 5) & 1;
    // pipeline: kick off the next tile's TDM before computing on this one
    if (j0 + 32 < NPAD)
      tdm_load_2d_f32(cur ? 0u : 2048u, biasBase + j0 + 32, 32, 16, NPAD, NPAD, NPAD);

    // ---- QK^T: B = k^T (K=dh contiguous in k's row-major (640,32)) ----
    Frag16 bk0, bk1;
#pragma unroll
    for (int vi = 0; vi < 8; ++vi) {
      const int kb = ((vi & 4) ? 16 : 0) + (half ? 8 : 0) + (vi & 3) * 2;
      bk0.u[vi] = *(const unsigned int*)(k + (bh * NPAD + j0 + nn) * DHEAD + kb);
      bk1.u[vi] = *(const unsigned int*)(k + (bh * NPAD + j0 + 16 + nn) * DHEAD + kb);
    }
    if (j0 + 32 < NPAD) {
      __builtin_prefetch(k + (bh * NPAD + j0 + 32 + nn) * DHEAD, 0, 1);
      __builtin_prefetch(vT + (bh * DHEAD + nn) * NPAD + j0 + 32, 0, 1);
    }
    v8f czero = {};
    v8f s0 = __builtin_amdgcn_wmma_f32_16x16x32_bf16(false, aq.v, false, bk0.v,
                                                     (short)0, czero, false, false);
    v8f s1 = __builtin_amdgcn_wmma_f32_16x16x32_bf16(false, aq.v, false, bk1.v,
                                                     (short)0, czero, false, false);

    // ---- wait for this iteration's bias tile (keep the next one in flight) ----
    if (j0 + 32 < NPAD) __builtin_amdgcn_s_wait_tensorcnt(1);
    else                __builtin_amdgcn_s_wait_tensorcnt(0);
    __asm__ volatile("" ::: "memory");
    // runtime select (no static pointer array: lld can't relocate LDS addrspacecast)
    const float* bt_ = (const float*)(ldsAll + (cur ? 512 : 0));

    // ---- bias + online softmax (row r+8*half lives across 16 lanes) ----
    float P0[8], P1[8];
#pragma unroll
    for (int r = 0; r < 8; ++r) {
      const int mm = r + 8 * half;
      float S0 = s0[r] + bt_[mm * 32 + nn];
      float S1 = s1[r] + bt_[mm * 32 + 16 + nn];
      float cmax = hmax16(fmaxf(S0, S1));
      const float nmax  = fmaxf(mrow[r], cmax);
      const float alpha = __expf(mrow[r] - nmax);
      P0[r] = __expf(S0 - nmax);
      P1[r] = __expf(S1 - nmax);
      const float rs = hsum16(P0[r] + P1[r]);
      lrow[r] = lrow[r] * alpha + rs;
      acc0[r] *= alpha;
      acc1[r] *= alpha;
      mrow[r] = nmax;
    }

    // ---- repack P (C layout -> A layout) through LDS ----
    unsigned short* ldsP = (unsigned short*)(ldsAll + 1024);
#pragma unroll
    for (int r = 0; r < 8; ++r) {
      const int mm = r + 8 * half;
      ldsP[mm * 32 + nn]      = f2bf(P0[r]);
      ldsP[mm * 32 + 16 + nn] = f2bf(P1[r]);
    }
    __syncthreads();
    Frag16 ap;
#pragma unroll
    for (int vi = 0; vi < 8; ++vi) {
      const int kb = ((vi & 4) ? 16 : 0) + (half ? 8 : 0) + (vi & 3) * 2;
      ap.u[vi] = *(const unsigned int*)(ldsP + m * 32 + kb);
    }
    __syncthreads();   // also guarantees bias ds-reads done before buffer reuse

    // ---- P @ V : B from vT (K=j contiguous) ----
    Frag16 bv0, bv1;
#pragma unroll
    for (int vi = 0; vi < 8; ++vi) {
      const int kb = ((vi & 4) ? 16 : 0) + (half ? 8 : 0) + (vi & 3) * 2;
      bv0.u[vi] = *(const unsigned int*)(vT + (bh * DHEAD + nn) * NPAD + j0 + kb);
      bv1.u[vi] = *(const unsigned int*)(vT + (bh * DHEAD + 16 + nn) * NPAD + j0 + kb);
    }
    acc0 = __builtin_amdgcn_wmma_f32_16x16x32_bf16(false, ap.v, false, bv0.v,
                                                   (short)0, acc0, false, false);
    acc1 = __builtin_amdgcn_wmma_f32_16x16x32_bf16(false, ap.v, false, bv1.v,
                                                   (short)0, acc1, false, false);
  }

  // ---- normalize and store to (b, 640, 256) bf16 ----
#pragma unroll
  for (int r = 0; r < 8; ++r) {
    const float inv = 1.0f / lrow[r];
    const int rowg = i0 + r + 8 * half;
    unsigned short* po = attnout + ((size_t)b * NPAD + rowg) * DMODEL + h * DHEAD;
    po[nn]      = f2bf(acc0[r] * inv);
    po[16 + nn] = f2bf(acc1[r] * inv);
  }
}

// =====================================================================
// Kernel 4: output projection + fused transpose store
// =====================================================================
__global__ __launch_bounds__(32)
void proj_kernel(const unsigned short* __restrict__ attnout,
                 const float* __restrict__ wout, float* __restrict__ out) {
  const int lane = threadIdx.x;
  const int ct = blockIdx.x;            // 0..15
  const int nt = blockIdx.y;            // 0..39
  const int b  = blockIdx.z;            // 0..63
  const int c0 = ct * 16, n0 = nt * 16;
  const int m    = lane & 15;
  const int half = lane >> 4;
  const int nn   = lane & 15;

  v8f acc = {};
#pragma unroll 2
  for (int k0 = 0; k0 < DMODEL; k0 += 32) {
    Frag16 a, bw;
#pragma unroll
    for (int vi = 0; vi < 8; ++vi) {
      const int kbA = ((vi & 4) ? 16 : 0) + (half ? 8 : 0) + (vi & 3) * 2;
      a.u[vi] = *(const unsigned int*)(attnout + ((size_t)b * NPAD + n0 + m) * DMODEL + k0 + kbA);
      const int kbB = ((vi & 4) ? 16 : 0) + (half ? 8 : 0) + (vi & 3) * 2;
      const float* pw = wout + (size_t)(k0 + kbB) * DMODEL + (c0 + nn);
      bw.u[vi] = pack_bf2(pw[0], pw[DMODEL]);
    }
    acc = __builtin_amdgcn_wmma_f32_16x16x32_bf16(false, a.v, false, bw.v,
                                                  (short)0, acc, false, false);
  }
#pragma unroll
  for (int r = 0; r < 8; ++r) {
    const int rowg = n0 + r + 8 * half;
    if (rowg < NWIN)
      out[((size_t)b * DMODEL + c0 + nn) * NWIN + rowg] = acc[r];
  }
}

// =====================================================================
extern "C" void kernel_launch(void* const* d_in, const int* in_sizes, int n_in,
                              void* d_out, int out_size, void* d_ws, size_t ws_size,
                              hipStream_t stream) {
  const float* x    = (const float*)d_in[0];
  const float* wqkv = (const float*)d_in[1];
  const float* wout = (const float*)d_in[2];
  const float* bt   = (const float*)d_in[3];
  const int*   rpi  = (const int*)d_in[4];
  float* out = (float*)d_out;

  // workspace layout (all L2-resident on MI455X: ~97 MB < 192 MB L2)
  const size_t E = (size_t)64 * NHEADS * NPAD * DHEAD;     // q/k/vT elems (bf16)
  unsigned short* q  = (unsigned short*)d_ws;
  unsigned short* k  = q + E;
  unsigned short* vT = k + E;
  float* biasPad = (float*)(vT + E);                        // 8*640*640 f32
  unsigned short* attnout =
      (unsigned short*)(biasPad + (size_t)NHEADS * NPAD * NPAD); // (64,640,256) bf16

  qkv_kernel<<<dim3(48, 40, 64), 32, 0, stream>>>(x, wqkv, q, k, vT);
  bias_kernel<<<(NHEADS * NPAD * NPAD + 255) / 256, 256, 0, stream>>>(bt, rpi, biasPad);
  attn_kernel<<<dim3(40, NHEADS, 64), 32, 0, stream>>>(q, k, vT, biasPad, attnout);
  proj_kernel<<<dim3(16, 40, 64), 32, 0, stream>>>(attnout, wout, out);
}
